// MultiHeadLatentAttention_59133109731521
// MI455X (gfx1250) — compile-verified, matrix operand order checked by reading, not run on hardware
//
#include <hip/hip_runtime.h>

// ---------------- problem constants ----------------
#define DIMX   2048
#define HN     16
#define NOPEX  128
#define ROPEX  64
#define QKDX   192      // NOPE + ROPE
#define VDX    128
#define QLRX   512
#define KVLRX  512
#define BBX    2
#define SSX    2048
#define MROWS  (BBX*SSX)   // 4096 token rows

typedef __bf16 bf16;
typedef __attribute__((ext_vector_type(16))) __bf16 bf16x16;
typedef __attribute__((ext_vector_type(8)))  float  f32x8;
typedef int v4i_vs __attribute__((__vector_size__(4 * sizeof(int))));

// =====================================================================
// Async global->LDS 16-byte copy (CDNA5 GLOBAL_LOAD_ASYNC_TO_LDS_B128,
// tracked on ASYNCcnt). Builtin when available, inline asm fallback.
// Builtin signature (from hipcc diagnostic): (v4i as1*, v4i as3*, Ii, Ii).
// =====================================================================
__device__ __forceinline__ void async_ld16(const void* gsrc, unsigned lds_off)
{
#if __has_builtin(__builtin_amdgcn_global_load_async_to_lds_b128)
    typedef __attribute__((address_space(1))) v4i_vs* gptr_t;
    typedef __attribute__((address_space(3))) v4i_vs* lptr_t;
    __builtin_amdgcn_global_load_async_to_lds_b128(
        (gptr_t)(unsigned long long)(uintptr_t)gsrc,
        (lptr_t)lds_off, 0, 0);
#else
    unsigned long long ga = (unsigned long long)(uintptr_t)gsrc;
    asm volatile("global_load_async_to_lds_b128 %0, %1, off"
                 :: "v"(lds_off), "v"(ga) : "memory");
#endif
}

__device__ __forceinline__ void wait_async0()
{
#if __has_builtin(__builtin_amdgcn_s_wait_asynccnt)
    __builtin_amdgcn_s_wait_asynccnt(0);
#else
    asm volatile("s_wait_asynccnt 0" ::: "memory");
#endif
}

// =====================================================================
// Generic GEMM: C[M,N] = A[M,K] * W[N,K]^T + bias[N]
// f32 inputs converted to bf16 in LDS, WMMA f32_16x16x32_bf16 accumulate.
// Block tile 128x64, 256 threads = 8 waves (4x2), each wave 32x32.
// Requires: M%128==0, N%64==0, K%32==0 (true for all call sites).
// =====================================================================
__global__ void __launch_bounds__(256)
gemm_bias_wmma(const float* __restrict__ A, const float* __restrict__ W,
               const float* __restrict__ bias, float* __restrict__ C,
               int M, int N, int K)
{
    __shared__ __align__(32) bf16 As[128][32];   // 8 KB
    __shared__ __align__(32) bf16 Bs[64][32];    // 4 KB

    const int tid  = threadIdx.x;
    const int lane = tid & 31;
    const int wid  = tid >> 5;     // 0..7
    const int wm   = wid >> 1;     // 0..3  (M direction)
    const int wn   = wid & 1;      // 0..1  (N direction)
    const int l16  = lane & 15;
    const int half = lane >> 4;    // 0/1
    const int bm   = blockIdx.x * 128;
    const int bn   = blockIdx.y * 64;

    f32x8 acc[2][2] = {};

    for (int k0 = 0; k0 < K; k0 += 32) {
        // ---- stage A tile (128x32): 2 threads/row, 16 floats each ----
        {
            const int row = tid >> 1;
            const int col = (tid & 1) * 16;
            const float* src = A + (size_t)(bm + row) * K + k0 + col;
            __builtin_prefetch(src + 32, 0, 0);   // global_prefetch next k-tile
            bf16* dst = &As[row][col];
#pragma unroll
            for (int i = 0; i < 16; i += 4) {
                float4 f = *(const float4*)(src + i);
                dst[i+0] = (bf16)f.x; dst[i+1] = (bf16)f.y;
                dst[i+2] = (bf16)f.z; dst[i+3] = (bf16)f.w;
            }
        }
        // ---- stage B tile (64x32): 4 threads/row, 8 floats each ----
        {
            const int row = tid >> 2;
            const int col = (tid & 3) * 8;
            const float* src = W + (size_t)(bn + row) * K + k0 + col;
            __builtin_prefetch(src + 32, 0, 0);
            bf16* dst = &Bs[row][col];
#pragma unroll
            for (int i = 0; i < 8; i += 4) {
                float4 f = *(const float4*)(src + i);
                dst[i+0] = (bf16)f.x; dst[i+1] = (bf16)f.y;
                dst[i+2] = (bf16)f.z; dst[i+3] = (bf16)f.w;
            }
        }
        __syncthreads();

        bf16x16 afrag[2], bfrag[2];
#pragma unroll
        for (int i = 0; i < 2; ++i)
            afrag[i] = *(const bf16x16*)&As[wm*32 + i*16 + l16][half*16];
#pragma unroll
        for (int j = 0; j < 2; ++j)
            bfrag[j] = *(const bf16x16*)&Bs[wn*32 + j*16 + l16][half*16];

#pragma unroll
        for (int i = 0; i < 2; ++i)
#pragma unroll
            for (int j = 0; j < 2; ++j)
                acc[i][j] = __builtin_amdgcn_wmma_f32_16x16x32_bf16(
                    false, afrag[i], false, bfrag[j],
                    (short)0, acc[i][j], false, false);
        __syncthreads();
    }

    // ---- epilogue: D layout lane: n = l16, rows m = r + half*8 ----
#pragma unroll
    for (int i = 0; i < 2; ++i) {
        const int mbase = bm + wm*32 + i*16 + half*8;
#pragma unroll
        for (int j = 0; j < 2; ++j) {
            const int n = bn + wn*32 + j*16 + l16;
            const float bv = bias[n];
#pragma unroll
            for (int r = 0; r < 8; ++r)
                C[(size_t)(mbase + r) * N + n] = acc[i][j][r] + bv;
        }
    }
}

// =====================================================================
// RMSNorm over 512 columns. One wave per row (8 rows / 256-thread block).
// =====================================================================
__global__ void __launch_bounds__(256)
rmsnorm512(const float* __restrict__ in, int in_stride,
           const float* __restrict__ w,
           float* __restrict__ out, int out_stride)
{
    const int lane = threadIdx.x & 31;
    const int row  = blockIdx.x * 8 + (threadIdx.x >> 5);
    const float* src = in + (size_t)row * in_stride;
    float v[16], ss = 0.f;
#pragma unroll
    for (int i = 0; i < 16; ++i) {
        v[i] = src[lane + i*32];
        ss += v[i] * v[i];
    }
#pragma unroll
    for (int o = 16; o >= 1; o >>= 1) ss += __shfl_xor(ss, o, 32);
    const float r = rsqrtf(ss * (1.0f/512.0f) + 1e-6f);
    float* dst = out + (size_t)row * out_stride;
#pragma unroll
    for (int i = 0; i < 16; ++i) {
        const int c = lane + i*32;
        dst[c] = v[i] * r * w[c];
    }
}

// =====================================================================
// RoPE for k_pe: kv_a[row, 512..575] -> k_pe[row, 0..63] (f32).
// =====================================================================
__global__ void __launch_bounds__(256)
rope_kpe(const float* __restrict__ kv_a, float* __restrict__ k_pe)
{
    const int lane = threadIdx.x & 31;
    const int row  = blockIdx.x * 8 + (threadIdx.x >> 5);
    const int s    = row & (SSX - 1);
    const float* src = kv_a + (size_t)row * (KVLRX + ROPEX) + KVLRX;
    const float x1 = src[2*lane], x2 = src[2*lane + 1];
    const float freq = __powf(10000.0f, -(float)lane * (1.0f/32.0f));
    float sn, cs;
    __sincosf((float)s * freq, &sn, &cs);
    k_pe[(size_t)row * ROPEX + 2*lane    ] = x1*cs - x2*sn;
    k_pe[(size_t)row * ROPEX + 2*lane + 1] = x1*sn + x2*cs;
}

// =====================================================================
// Pack Q: q_b[row, h*192..] (f32) -> q_full[b,h,s,192] (bf16), RoPE last 64.
// =====================================================================
__global__ void __launch_bounds__(256)
pack_q(const float* __restrict__ q_b, bf16* __restrict__ q_full)
{
    const int gw   = (blockIdx.x * 256 + threadIdx.x) >> 5;
    const int lane = threadIdx.x & 31;
    const int h    = gw & (HN - 1);
    const int row  = gw >> 4;               // b*S + s
    const int b    = row >> 11;             // /SSX
    const int s    = row & (SSX - 1);
    const float* src = q_b + (size_t)row * (HN*QKDX) + h*QKDX;
    bf16* dst = q_full + (((size_t)(b*HN + h) * SSX) + s) * QKDX;
#pragma unroll
    for (int i = 0; i < 4; ++i) {
        const int c = lane + i*32;
        dst[c] = (bf16)src[c];
    }
    const float x1 = src[NOPEX + 2*lane], x2 = src[NOPEX + 2*lane + 1];
    const float freq = __powf(10000.0f, -(float)lane * (1.0f/32.0f));
    float sn, cs;
    __sincosf((float)s * freq, &sn, &cs);
    dst[NOPEX + 2*lane    ] = (bf16)(x1*cs - x2*sn);
    dst[NOPEX + 2*lane + 1] = (bf16)(x1*sn + x2*cs);
}

// =====================================================================
// Pack K & V: kv_b[row, h*256..] + k_pe[row] ->
//   k_full[b,h,s,192] bf16, vT[b,h,v,s] bf16 (V transposed for WMMA B-frags)
// =====================================================================
__global__ void __launch_bounds__(256)
pack_kv(const float* __restrict__ kv_b, const float* __restrict__ k_pe,
        bf16* __restrict__ k_full, bf16* __restrict__ vT)
{
    const int gw   = (blockIdx.x * 256 + threadIdx.x) >> 5;
    const int lane = threadIdx.x & 31;
    const int h    = gw & (HN - 1);
    const int row  = gw >> 4;
    const int b    = row >> 11;
    const int s    = row & (SSX - 1);
    const float* src = kv_b + (size_t)row * (HN*(NOPEX+VDX)) + h*(NOPEX+VDX);
    bf16* kd = k_full + (((size_t)(b*HN + h) * SSX) + s) * QKDX;
#pragma unroll
    for (int i = 0; i < 4; ++i) {
        const int c = lane + i*32;
        kd[c] = (bf16)src[c];
    }
    const float* kp = k_pe + (size_t)row * ROPEX;
#pragma unroll
    for (int i = 0; i < 2; ++i) {
        const int c = lane + i*32;
        kd[NOPEX + c] = (bf16)kp[c];
    }
    bf16* vd = vT + (size_t)(b*HN + h) * VDX * SSX;
#pragma unroll
    for (int i = 0; i < 4; ++i) {
        const int v = lane + i*32;
        vd[(size_t)v * SSX + s] = (bf16)src[NOPEX + v];
    }
}

// =====================================================================
// Flash attention with async-LDS K/V staging.
// 256 threads = 8 waves; each wave owns 16 query rows. The block
// cooperatively stages each 32-key tile of K (32x192 bf16, contiguous)
// and V^T (128x32 bf16) into LDS via GLOBAL_LOAD_ASYNC_TO_LDS_B128,
// synchronized with s_wait_asynccnt + workgroup barrier. Waves whose
// causal horizon is below the tile skip compute (wave-uniform branch,
// EXEC stays all-ones around WMMA).
// =====================================================================
__global__ void __launch_bounds__(256)
flash_mla(const bf16* __restrict__ q_full, const bf16* __restrict__ k_full,
          const bf16* __restrict__ vT, float* __restrict__ out)
{
    __shared__ __align__(32) bf16 Ks[32][QKDX];   // 12 KB: 32 keys x 192 dims
    __shared__ __align__(32) bf16 Vs[VDX][32];    //  8 KB: 128 vdims x 32 keys

    const int tid  = threadIdx.x;
    const int lane = tid & 31;
    const int wid  = tid >> 5;
    const int l16  = lane & 15;
    const int half = lane >> 4;
    const int bh   = blockIdx.y;              // b*16 + h
    const int b    = bh >> 4, h = bh & 15;
    const int qbase= blockIdx.x * 128;
    const int qm   = qbase + wid * 16;        // wave's first query row

    const bf16* qrows = q_full + ((size_t)bh * SSX + qm) * QKDX;
    const bf16* kbase = k_full + (size_t)bh * SSX * QKDX;
    const bf16* vbase = vT     + (size_t)bh * VDX * SSX;

    const unsigned ks_lds = (unsigned)(uintptr_t)&Ks[0][0];
    const unsigned vs_lds = (unsigned)(uintptr_t)&Vs[0][0];

    // Q as B-fragments: lane holds query n=l16, dims f*32 + half*16 .. +15
    bf16x16 qf[6];
#pragma unroll
    for (int f = 0; f < 6; ++f)
        qf[f] = *(const bf16x16*)(qrows + (size_t)l16 * QKDX + f*32 + half*16);

    f32x8 acc[8] = {};
    float m_run = -1e30f, l_run = 0.f;
    const float scale = 0.0721687836487032f;   // 1/sqrt(192)
    const int qg = qm + l16;                   // this lane's query (as S^T col)
    const int NT = qbase / 32 + 4;             // tiles needed by whole block

    for (int t = 0; t < NT; ++t) {
        const int kn = t * 32;

        // ---- async stage: K tile is one contiguous 12288-B block ----
        {
            const bf16* gk = kbase + (size_t)kn * QKDX;
#pragma unroll
            for (int c = 0; c < 3; ++c) {
                const int chunk = tid + c * 256;          // 0..767 x 16 B
                async_ld16(gk + chunk * 8, ks_lds + chunk * 16);
            }
#pragma unroll
            for (int c = 0; c < 2; ++c) {
                const int chunk = tid + c * 256;          // 0..511 x 16 B
                const int v = chunk >> 2, part = chunk & 3;
                async_ld16(vbase + (size_t)v * SSX + kn + part * 8,
                           vs_lds + chunk * 16);
            }
        }
        wait_async0();
        __syncthreads();

        if (kn <= qm + 15) {   // wave-uniform causal skip
            // ---- scores transposed: rows = keys, cols = queries ----
            f32x8 st0 = {}, st1 = {};
#pragma unroll
            for (int f = 0; f < 6; ++f) {
                bf16x16 ka = *(const bf16x16*)&Ks[l16][f*32 + half*16];
                st0 = __builtin_amdgcn_wmma_f32_16x16x32_bf16(
                    false, ka, false, qf[f], (short)0, st0, false, false);
            }
#pragma unroll
            for (int f = 0; f < 6; ++f) {
                bf16x16 ka = *(const bf16x16*)&Ks[16 + l16][f*32 + half*16];
                st1 = __builtin_amdgcn_wmma_f32_16x16x32_bf16(
                    false, ka, false, qf[f], (short)0, st1, false, false);
            }
            // ---- scale + causal mask; key = kn + half*8 + r (+16 st1) ----
            float s0[8], s1[8];
#pragma unroll
            for (int r = 0; r < 8; ++r) {
                const int k0 = kn + half*8 + r;
                s0[r] = (k0      <= qg) ? st0[r]*scale : -1e30f;
                s1[r] = (k0 + 16 <= qg) ? st1[r]*scale : -1e30f;
            }
            // ---- per-query max over 32 keys ----
            float mx = s0[0];
#pragma unroll
            for (int r = 0; r < 8; ++r) { mx = fmaxf(mx, s0[r]); mx = fmaxf(mx, s1[r]); }
            mx = fmaxf(mx, __shfl_xor(mx, 16, 32));
            const float m_new = fmaxf(m_run, mx);
            float p0[8], p1[8], rs = 0.f;
#pragma unroll
            for (int r = 0; r < 8; ++r) {
                p0[r] = __expf(s0[r] - m_new);
                p1[r] = __expf(s1[r] - m_new);
                rs += p0[r] + p1[r];
            }
            rs += __shfl_xor(rs, 16, 32);
            const float alpha = __expf(m_run - m_new);
            l_run = l_run * alpha + rs;
            m_run = m_new;
            // ---- rescale O accumulators (alpha per query via bpermute) ----
#pragma unroll
            for (int r = 0; r < 8; ++r) {
                const float ar = __shfl(alpha, r + half*8, 32);
#pragma unroll
                for (int j = 0; j < 8; ++j) acc[j][r] *= ar;
            }
            // ---- repack P into A-fragment: lane = query row, K = 32 keys ----
            float pp0[8], pp1[8];
#pragma unroll
            for (int r = 0; r < 8; ++r) {
                pp0[r] = __shfl_xor(p0[r], 16, 32);
                pp1[r] = __shfl_xor(p1[r], 16, 32);
            }
            bf16x16 pf;
#pragma unroll
            for (int kk = 0; kk < 16; ++kk) {
                float v;
                if (kk < 8) v = half ? pp1[kk]  : p0[kk];
                else        v = half ? p1[kk-8] : pp0[kk-8];
                pf[kk] = (bf16)v;
            }
            // ---- O += P · V (V^T rows contiguous in LDS per lane) ----
#pragma unroll
            for (int j = 0; j < 8; ++j) {
                bf16x16 vf = *(const bf16x16*)&Vs[j*16 + l16][half*16];
                acc[j] = __builtin_amdgcn_wmma_f32_16x16x32_bf16(
                    false, pf, false, vf, (short)0, acc[j], false, false);
            }
        }
        __syncthreads();
    }

    // ---- finalize: divide by l, store to [b, s, h*128 + v] ----
    const float inv_l = 1.0f / l_run;
#pragma unroll
    for (int r = 0; r < 8; ++r) {
        const float il = __shfl(inv_l, r + half*8, 32);
        const int qrow = qm + r + half*8;
        const size_t base = ((size_t)b * SSX + qrow) * (HN*VDX) + h*VDX;
#pragma unroll
        for (int j = 0; j < 8; ++j)
            out[base + j*16 + l16] = acc[j][r] * il;
    }
}

// =====================================================================
// Launcher
// =====================================================================
extern "C" void kernel_launch(void* const* d_in, const int* in_sizes, int n_in,
                              void* d_out, int out_size, void* d_ws, size_t ws_size,
                              hipStream_t stream)
{
    (void)in_sizes; (void)n_in; (void)out_size; (void)ws_size;
    const float* x        = (const float*)d_in[0];
    /* d_in[1] = mask (int32) unused: causal mask applied analytically */
    const float* wq_a_w   = (const float*)d_in[2];
    const float* wq_a_b   = (const float*)d_in[3];
    const float* q_norm_w = (const float*)d_in[4];
    const float* wq_b_w   = (const float*)d_in[5];
    const float* wq_b_b   = (const float*)d_in[6];
    const float* wkv_a_w  = (const float*)d_in[7];
    const float* wkv_a_b  = (const float*)d_in[8];
    const float* kv_norm_w= (const float*)d_in[9];
    const float* wkv_b_w  = (const float*)d_in[10];
    const float* wkv_b_b  = (const float*)d_in[11];
    const float* wo_w     = (const float*)d_in[12];
    const float* wo_b     = (const float*)d_in[13];
    float* out = (float*)d_out;

    // ---- workspace arena (lifetime-based reuse of the big region) ----
    char* ws = (char*)d_ws;
    size_t off = 0;
    auto arena = [&](size_t bytes) -> void* {
        void* p = ws + off;
        off += (bytes + 255) & ~(size_t)255;
        return p;
    };
    float* q_a     = (float*)arena((size_t)MROWS * QLRX * 4);          //  8 MiB
    float* kv_a    = (float*)arena((size_t)MROWS * (KVLRX+ROPEX) * 4); //  9 MiB
    float* kv_nrm  = (float*)arena((size_t)MROWS * KVLRX * 4);         //  8 MiB
    float* k_pe    = (float*)arena((size_t)MROWS * ROPEX * 4);         //  1 MiB
    bf16*  q_full  = (bf16*)arena((size_t)BBX*HN*SSX*QKDX * 2);        // 24 MiB
    bf16*  k_full  = (bf16*)arena((size_t)BBX*HN*SSX*QKDX * 2);        // 24 MiB
    bf16*  vT      = (bf16*)arena((size_t)BBX*HN*VDX*SSX * 2);         // 16 MiB
    void*  big     = arena((size_t)MROWS * (HN*(NOPEX+VDX)) * 4);      // 64 MiB shared
    float* q_b      = (float*)big;   // lives: gemm2 -> pack_q
    float* kv_b     = (float*)big;   // lives: gemm4 -> pack_kv
    float* attn_out = (float*)big;   // lives: flash -> gemm5

    const dim3 blk256(256);

    // ---------------- Q path ----------------
    gemm_bias_wmma<<<dim3(MROWS/128, QLRX/64), blk256, 0, stream>>>(
        x, wq_a_w, wq_a_b, q_a, MROWS, QLRX, DIMX);
    rmsnorm512<<<dim3(MROWS/8), blk256, 0, stream>>>(
        q_a, QLRX, q_norm_w, q_a, QLRX);                      // in-place
    gemm_bias_wmma<<<dim3(MROWS/128, (HN*QKDX)/64), blk256, 0, stream>>>(
        q_a, wq_b_w, wq_b_b, q_b, MROWS, HN*QKDX, QLRX);
    pack_q<<<dim3(MROWS*HN/8), blk256, 0, stream>>>(q_b, q_full);

    // ---------------- KV path ----------------
    gemm_bias_wmma<<<dim3(MROWS/128, (KVLRX+ROPEX)/64), blk256, 0, stream>>>(
        x, wkv_a_w, wkv_a_b, kv_a, MROWS, KVLRX+ROPEX, DIMX);
    rmsnorm512<<<dim3(MROWS/8), blk256, 0, stream>>>(
        kv_a, KVLRX+ROPEX, kv_norm_w, kv_nrm, KVLRX);
    rope_kpe<<<dim3(MROWS/8), blk256, 0, stream>>>(kv_a, k_pe);
    gemm_bias_wmma<<<dim3(MROWS/128, (HN*(NOPEX+VDX))/64), blk256, 0, stream>>>(
        kv_nrm, wkv_b_w, wkv_b_b, kv_b, MROWS, HN*(NOPEX+VDX), KVLRX);
    pack_kv<<<dim3(MROWS*HN/8), blk256, 0, stream>>>(kv_b, k_pe, k_full, vT);

    // ---------------- attention (flash, causal, async-LDS staged) ----------------
    flash_mla<<<dim3(SSX/128, BBX*HN), blk256, 0, stream>>>(
        q_full, k_full, vT, attn_out);

    // ---------------- output projection ----------------
    gemm_bias_wmma<<<dim3(MROWS/128, DIMX/64), blk256, 0, stream>>>(
        attn_out, wo_w, wo_b, out, MROWS, DIMX, HN*VDX);
}
